// KernelProjectionT2IAttn_55267639165045
// MI455X (gfx1250) — compile-verified, matrix operand order checked by reading, not run on hardware
//
#include <hip/hip_runtime.h>
#include <hip/hip_bf16.h>

// ---------------------------------------------------------------------------
// Problem constants (from the reference)
// ---------------------------------------------------------------------------
#define BN 4096   // batch
#define LN 1024   // embed dim L
#define LR 256    // L / K
#define RN 36     // img rows summed
#define TN 32     // caption positions (only row 0 used)

typedef __bf16 bf16_t;
typedef __attribute__((ext_vector_type(16))) __bf16 v16bf;
typedef __attribute__((ext_vector_type(8)))  __bf16 v8bf;
typedef __attribute__((ext_vector_type(8)))  float  v8f;

// ---------------------------------------------------------------------------
// WMMA fragment loader for "X @ Y.T" (NT) GEMMs with row-major X and Y.
//
// A-matrix 16x32 bf16 per ISA 7.12.2: lanes 0-15 hold row M=lane,
// K = {k0..k0+7} in v0-3 and {k0+16..k0+23} in v4-7; lanes 16-31 hold the
// same rows with K = {k0+8..k0+15, k0+24..k0+31}.  The B-matrix (32x16)
// layout is the lane-symmetric transpose: lane n holds column n of B_wmma,
// i.e. row n of Y when computing X @ Y.T.  Hence one loader serves both:
// lane L loads 2 x 16B from (row0 + (L&15))*ld + k0 + 8*(L>>4)  (+16).
// ---------------------------------------------------------------------------
__device__ __forceinline__ v16bf load_frag_nt(const bf16_t* __restrict__ mat,
                                              int ld, int row0, int k0, int lane) {
    const int m  = lane & 15;
    const int kb = k0 + ((lane >> 4) << 3);
    const bf16_t* p = mat + (size_t)(row0 + m) * ld + kb;
    union { v16bf f; v8bf h[2]; } u;
    u.h[0] = *(const v8bf*)(p);
    u.h[1] = *(const v8bf*)(p + 16);
    return u.f;
}

// 16x16 f32 C/D tile layout: lane -> column (lane&15), VGPR r -> row r + 8*(lane>=16)

// ---------------------------------------------------------------------------
// Kernel 1: f32 -> bf16 converter (weights)
// ---------------------------------------------------------------------------
__global__ void cvt_f32_bf16(const float* __restrict__ in, bf16_t* __restrict__ out, int n) {
    int i = blockIdx.x * blockDim.x + threadIdx.x;
    if (i < n) out[i] = (bf16_t)in[i];
}

// ---------------------------------------------------------------------------
// Kernel 2: per-batch-row prep.
//   s[b,:]      = sum_r img[b,r,:]        -> A1 = bf16(s), A2 = bf16(s*s)
//   cap0[b,:]   = cap[b,0,:]              -> C0h = bf16(cap0)
//   inv_norm[b] = rsqrt(sum cap0^2)
// grid = 4096 blocks, 256 threads
// ---------------------------------------------------------------------------
__global__ __launch_bounds__(256)
void prep_rows(const float* __restrict__ img, const float* __restrict__ cap,
               bf16_t* __restrict__ A1, bf16_t* __restrict__ A2,
               bf16_t* __restrict__ C0h, float* __restrict__ inv_norm) {
    const int b = blockIdx.x;
    const int t = threadIdx.x;
    __shared__ float red[256];

    float nacc = 0.f;
    #pragma unroll
    for (int li = 0; li < LN / 256; ++li) {
        const int l = t + li * 256;
        const float* ip = img + ((size_t)b * RN) * LN + l;
        float sv = 0.f;
        #pragma unroll 6
        for (int r = 0; r < RN; ++r) sv += ip[(size_t)r * LN];
        A1[(size_t)b * LN + l] = (bf16_t)sv;
        A2[(size_t)b * LN + l] = (bf16_t)(sv * sv);
        const float c = cap[((size_t)b * TN) * LN + l];
        C0h[(size_t)b * LN + l] = (bf16_t)c;
        nacc += c * c;
    }
    red[t] = nacc;
    __syncthreads();
    #pragma unroll
    for (int off = 128; off > 0; off >>= 1) {
        if (t < off) red[t] += red[t + off];
        __syncthreads();
    }
    if (t == 0) inv_norm[b] = rsqrtf(red[0]);
}

// ---------------------------------------------------------------------------
// Kernel 3: cap_repr = bf16( cap0 @ W_red.T + b_red )   (M=4096, N=256, K=1024)
// Block tile 128(M) x 64(N), 8 waves (4 along M x 2 along N), wave tile 32x32.
// ---------------------------------------------------------------------------
template <int KDIM>
__global__ __launch_bounds__(256)
void gemm_bias_bf16(const bf16_t* __restrict__ A, const bf16_t* __restrict__ Bm,
                    const float* __restrict__ bias, bf16_t* __restrict__ outh, int N) {
    const int lane = threadIdx.x & 31;
    const int wave = threadIdx.x >> 5;
    const int row0 = blockIdx.y * 128 + (wave & 3) * 32;
    const int col0 = blockIdx.x * 64  + (wave >> 2) * 32;

    v8f acc[2][2] = {};
    #pragma unroll 2
    for (int k0 = 0; k0 < KDIM; k0 += 32) {
        v16bf a[2], bfr[2];
        #pragma unroll
        for (int i = 0; i < 2; ++i) a[i]   = load_frag_nt(A,  KDIM, row0 + 16 * i, k0, lane);
        #pragma unroll
        for (int j = 0; j < 2; ++j) bfr[j] = load_frag_nt(Bm, KDIM, col0 + 16 * j, k0, lane);
        #pragma unroll
        for (int i = 0; i < 2; ++i)
            #pragma unroll
            for (int j = 0; j < 2; ++j)
                acc[i][j] = __builtin_amdgcn_wmma_f32_16x16x32_bf16(
                    false, a[i], false, bfr[j], (short)0, acc[i][j], false, false);
    }

    const int n   = lane & 15;
    const int mhi = (lane >> 4) * 8;
    #pragma unroll
    for (int i = 0; i < 2; ++i)
        #pragma unroll
        for (int j = 0; j < 2; ++j) {
            const int col = col0 + 16 * j + n;
            const float bv = bias[col];
            bf16_t* p = outh + (size_t)(row0 + 16 * i + mhi) * N + col;
            #pragma unroll
            for (int r = 0; r < 8; ++r) p[(size_t)r * N] = (bf16_t)(acc[i][j][r] + bv);
        }
}

// ---------------------------------------------------------------------------
// Kernel 4: g = sigmoid( cap_repr @ W_kp.T + b_kp )   (M=4096, N=1024, K=256)
// Epilogue: B1 = bf16(g * cap0 * inv_norm), B2 = bf16(g * g)
// ---------------------------------------------------------------------------
template <int KDIM>
__global__ __launch_bounds__(256)
void gate_kernel(const bf16_t* __restrict__ A, const bf16_t* __restrict__ Bm,
                 const float* __restrict__ bkp, const bf16_t* __restrict__ c0h,
                 const float* __restrict__ invn,
                 bf16_t* __restrict__ B1, bf16_t* __restrict__ B2, int N) {
    const int lane = threadIdx.x & 31;
    const int wave = threadIdx.x >> 5;
    const int row0 = blockIdx.y * 128 + (wave & 3) * 32;
    const int col0 = blockIdx.x * 64  + (wave >> 2) * 32;

    v8f acc[2][2] = {};
    #pragma unroll
    for (int k0 = 0; k0 < KDIM; k0 += 32) {
        v16bf a[2], bfr[2];
        #pragma unroll
        for (int i = 0; i < 2; ++i) a[i]   = load_frag_nt(A,  KDIM, row0 + 16 * i, k0, lane);
        #pragma unroll
        for (int j = 0; j < 2; ++j) bfr[j] = load_frag_nt(Bm, KDIM, col0 + 16 * j, k0, lane);
        #pragma unroll
        for (int i = 0; i < 2; ++i)
            #pragma unroll
            for (int j = 0; j < 2; ++j)
                acc[i][j] = __builtin_amdgcn_wmma_f32_16x16x32_bf16(
                    false, a[i], false, bfr[j], (short)0, acc[i][j], false, false);
    }

    const int n   = lane & 15;
    const int mhi = (lane >> 4) * 8;
    #pragma unroll
    for (int i = 0; i < 2; ++i)
        #pragma unroll
        for (int j = 0; j < 2; ++j) {
            const int col = col0 + 16 * j + n;       // L index
            const float bk = bkp[col];
            #pragma unroll
            for (int r = 0; r < 8; ++r) {
                const int m = row0 + 16 * i + mhi + r;   // batch row
                const float pre = acc[i][j][r] + bk;
                const float g   = 1.0f / (1.0f + __expf(-pre));
                const float ch  = (float)c0h[(size_t)m * LN + col] * invn[m];
                B1[(size_t)m * LN + col] = (bf16_t)(g * ch);
                B2[(size_t)m * LN + col] = (bf16_t)(g * g);
            }
        }
}

// ---------------------------------------------------------------------------
// Kernel 5: fused dual GEMM (M=N=4096, K=1024):
//   numer   = A1 @ B1.T       denomsq = A2 @ B2.T
//   out     = numer * rsqrt(denomsq)
// ---------------------------------------------------------------------------
template <int KDIM>
__global__ __launch_bounds__(256)
void dual_gemm(const bf16_t* __restrict__ A1, const bf16_t* __restrict__ A2,
               const bf16_t* __restrict__ B1, const bf16_t* __restrict__ B2,
               float* __restrict__ out, int N) {
    const int lane = threadIdx.x & 31;
    const int wave = threadIdx.x >> 5;
    const int row0 = blockIdx.y * 128 + (wave & 3) * 32;
    const int col0 = blockIdx.x * 64  + (wave >> 2) * 32;

    v8f acc1[2][2] = {};
    v8f acc2[2][2] = {};
    #pragma unroll 2
    for (int k0 = 0; k0 < KDIM; k0 += 32) {
        v16bf a1[2], a2[2], b1[2], b2[2];
        #pragma unroll
        for (int i = 0; i < 2; ++i) {
            a1[i] = load_frag_nt(A1, KDIM, row0 + 16 * i, k0, lane);
            a2[i] = load_frag_nt(A2, KDIM, row0 + 16 * i, k0, lane);
        }
        #pragma unroll
        for (int j = 0; j < 2; ++j) {
            b1[j] = load_frag_nt(B1, KDIM, col0 + 16 * j, k0, lane);
            b2[j] = load_frag_nt(B2, KDIM, col0 + 16 * j, k0, lane);
        }
        #pragma unroll
        for (int i = 0; i < 2; ++i)
            #pragma unroll
            for (int j = 0; j < 2; ++j) {
                acc1[i][j] = __builtin_amdgcn_wmma_f32_16x16x32_bf16(
                    false, a1[i], false, b1[j], (short)0, acc1[i][j], false, false);
                acc2[i][j] = __builtin_amdgcn_wmma_f32_16x16x32_bf16(
                    false, a2[i], false, b2[j], (short)0, acc2[i][j], false, false);
            }
    }

    const int n   = lane & 15;
    const int mhi = (lane >> 4) * 8;
    #pragma unroll
    for (int i = 0; i < 2; ++i)
        #pragma unroll
        for (int j = 0; j < 2; ++j) {
            float* p = out + (size_t)(row0 + 16 * i + mhi) * N + col0 + 16 * j + n;
            #pragma unroll
            for (int r = 0; r < 8; ++r)
                p[(size_t)r * N] = acc1[i][j][r] * rsqrtf(acc2[i][j][r]);
        }
}

// ---------------------------------------------------------------------------
// Host side
// ---------------------------------------------------------------------------
extern "C" void kernel_launch(void* const* d_in, const int* in_sizes, int n_in,
                              void* d_out, int out_size, void* d_ws, size_t ws_size,
                              hipStream_t stream) {
    const float* img   = (const float*)d_in[0];   // (B, R, L)
    const float* cap   = (const float*)d_in[1];   // (B, T, L)
    /* d_in[2] = lens: unused by the reference */
    const float* W_red = (const float*)d_in[3];   // (L/K, L)
    const float* b_red = (const float*)d_in[4];   // (L/K,)
    const float* W_kp  = (const float*)d_in[5];   // (L, L/K)
    const float* b_kp  = (const float*)d_in[6];   // (L,)
    float* out = (float*)d_out;                   // (B, B) f32

    // Workspace layout (bf16 matrices, all 16B-aligned)
    const size_t SB = (size_t)BN * LN * sizeof(bf16_t);  // 8 MiB
    char* ws = (char*)d_ws;
    bf16_t* A1h  = (bf16_t*)(ws);                         // bf16(s)
    bf16_t* A2h  = (bf16_t*)(ws + SB);                    // bf16(s*s)
    bf16_t* C0h  = (bf16_t*)(ws + 2 * SB);                // bf16(cap0)
    bf16_t* B1h  = (bf16_t*)(ws + 3 * SB);                // bf16(g * c_hat)
    bf16_t* B2h  = (bf16_t*)(ws + 4 * SB);                // bf16(g * g)
    bf16_t* CRh  = (bf16_t*)(ws + 5 * SB);                // cap_repr bf16 (2 MiB)
    bf16_t* WRh  = (bf16_t*)(ws + 5 * SB + (size_t)BN * LR * 2);
    bf16_t* WKh  = (bf16_t*)(ws + 5 * SB + (size_t)BN * LR * 2 + (size_t)LR * LN * 2);
    float*  invn = (float*) (ws + 5 * SB + (size_t)BN * LR * 2 + 2 * (size_t)LR * LN * 2);

    // 1) weights -> bf16
    cvt_f32_bf16<<<(LR * LN + 255) / 256, 256, 0, stream>>>(W_red, WRh, LR * LN);
    cvt_f32_bf16<<<(LN * LR + 255) / 256, 256, 0, stream>>>(W_kp,  WKh, LN * LR);

    // 2) row prep: s, s^2, cap0, inv_norm
    prep_rows<<<BN, 256, 0, stream>>>(img, cap, A1h, A2h, C0h, invn);

    // 3) cap_repr = cap0 @ W_red.T + b_red   (4096 x 256, K=1024)
    gemm_bias_bf16<LN><<<dim3(LR / 64, BN / 128), 256, 0, stream>>>(C0h, WRh, b_red, CRh, LR);

    // 4) gate + B1/B2 operands              (4096 x 1024, K=256)
    gate_kernel<LR><<<dim3(LN / 64, BN / 128), 256, 0, stream>>>(CRh, WKh, b_kp, C0h, invn, B1h, B2h, LN);

    // 5) fused numer/denom dual GEMM        (4096 x 4096, K=1024)
    dual_gemm<LN><<<dim3(BN / 64, BN / 128), 256, 0, stream>>>(A1h, A2h, B1h, B2h, out, BN);
}
